// gnn_encoder_19327352832219
// MI455X (gfx1250) — matrix operand order repaired and checked by reading, not compiled, not run on hardware
//
#include <hip/hip_runtime.h>
#include <hip/hip_bf16.h>

#define NNODE 8192
#define NEDGE 131072
#define NGRAPH 64
#define FE 32

typedef __attribute__((ext_vector_type(16))) __bf16 v16bf;
typedef __attribute__((ext_vector_type(8)))  float  v8f;
typedef __attribute__((ext_vector_type(4)))  unsigned v4u;
typedef __attribute__((ext_vector_type(8)))  int   v8i;
typedef __attribute__((ext_vector_type(4)))  int   v4i;

union FragAB { v16bf v; unsigned u[8]; };

__device__ __forceinline__ unsigned short f2bf(float f) {
  unsigned u = __float_as_uint(f);
  unsigned r = u + 0x7FFFu + ((u >> 16) & 1u);
  return (unsigned short)(r >> 16);
}
__device__ __forceinline__ float bf2f(unsigned short h) {
  return __uint_as_float(((unsigned)h) << 16);
}
// CDNA5 packed fp32->bf16 conversion (RTNE)
__device__ __forceinline__ unsigned pk2bf(float lo, float hi) {
  unsigned d;
  asm("v_cvt_pk_bf16_f32 %0, %1, %2" : "=v"(d) : "v"(lo), "v"(hi));
  return d;
}
// CDNA5 VOP3P packed bf16 multiply (rank-1 A-fragment scaling)
__device__ __forceinline__ unsigned pk_mul_bf16(unsigned a, unsigned b) {
  unsigned d;
  asm volatile("v_pk_mul_bf16 %0, %1, %2" : "=v"(d) : "v"(a), "v"(b));
  return d;
}
__device__ __forceinline__ v8f wmma_bf16(const unsigned* a8, const unsigned* b8, v8f c) {
  FragAB a, b;
#pragma unroll
  for (int i = 0; i < 8; ++i) { a.u[i] = a8[i]; b.u[i] = b8[i]; }
  return __builtin_amdgcn_wmma_f32_16x16x32_bf16(false, a.v, false, b.v, (short)0, c,
                                                 false, false);
}

// Tensor Data Mover: async DMA of a 2-D bf16 tile (rows x tile_cols, contiguous LDS fill).
// D# per CDNA5 ISA 8.3/8.4: group0 = {count|flags, lds_addr, global_addr, type=2},
// group1 = {data_size=2B, tensor_dim0/1, tile_dim0/1, tensor_dim0_stride}.
// This toolchain's builtin takes 6 args (g0, g1, g2, g3, g4-spare, cpol).
__device__ __forceinline__ void tdm_load_2d_bf16(unsigned lds_off, const void* gaddr,
                                                 unsigned rows, unsigned row_elems,
                                                 unsigned tile_cols,
                                                 unsigned row_stride_elems) {
  unsigned long long ga = (unsigned long long)(uintptr_t)gaddr;
  v4u g0;
  g0.x = 1u;                                                   // count=1 (valid user D#)
  g0.y = lds_off;                                              // LDS byte address
  g0.z = (unsigned)(ga & 0xFFFFFFFFu);                         // global_addr[31:0]
  g0.w = (unsigned)((ga >> 32) & 0x01FFFFFFu) | 0x80000000u;   // addr[56:32] | type=2
  v8i g1;
  g1[0] = (int)(1u << 16);                                     // data_size=2B
  g1[1] = (int)((row_elems & 0xFFFFu) << 16);                  // tensor_dim0[15:0]
  g1[2] = (int)((row_elems >> 16) | ((rows & 0xFFFFu) << 16)); // dim0[31:16]|dim1[15:0]
  g1[3] = (int)((rows >> 16) | (tile_cols << 16));             // dim1[31:16]|tile_dim0
  g1[4] = (int)(rows & 0xFFFFu);                               // tile_dim1 (tile_dim2=0)
  g1[5] = (int)row_stride_elems;                               // tensor_dim0_stride[31:0]
  g1[6] = 0;
  g1[7] = 0;
  v4i z4 = {};
  v8i z8 = {};
  __builtin_amdgcn_tensor_load_to_lds(g0, g1, z4, z4, z8, 0);
}

// A-fragment (16x32 bf16) from LDS row-major bf16 tile; ISA 7.12.2 16-bit A layout:
// lane L: row m=L&15; K set = {0..7,16..23} (L<16) or {8..15,24..31} (L>=16)
__device__ __forceinline__ void a_frag_lds(const unsigned short* base, int ldm, int kbase,
                                           unsigned* a) {
  int lane = threadIdx.x & 31;
  int m = lane & 15, khi = (lane >> 4) * 8;
  const unsigned short* row = base + m * ldm + kbase + khi;
#pragma unroll
  for (int v = 0; v < 8; ++v) {
    int k = ((v < 4) ? 0 : 16) + (v & 3) * 2;
    a[v] = *(const unsigned*)(row + k);
  }
}
// A-fragment built directly from global fp32 rows (packed-converted to bf16 pairs)
__device__ __forceinline__ void a_frag_gf32(const float* base, int ldm, int kbase,
                                            unsigned* a) {
  int lane = threadIdx.x & 31;
  int m = lane & 15, khi = (lane >> 4) * 8;
  const float* row = base + m * ldm + kbase + khi;
#pragma unroll
  for (int v = 0; v < 8; ++v) {
    int k = ((v < 4) ? 0 : 16) + (v & 3) * 2;
    a[v] = pk2bf(row[k], row[k + 1]);
  }
}
// B-fragment (32x16 bf16) from LDS staged N-major [n][K]; lane holds col n=L&15,
// K set = {0..7,16..23} (L<16) or {8..15,24..31} (L>=16), matching the A layout
__device__ __forceinline__ void b_frag_lds(const unsigned short* base, int ldk, int kbase,
                                           unsigned* b) {
  int lane = threadIdx.x & 31;
  int n = lane & 15, kb = kbase + (lane >> 4) * 8;
  const unsigned* p = (const unsigned*)(base + n * ldk + kb);
#pragma unroll
  for (int v = 0; v < 8; ++v) b[v] = p[(v < 4) ? v : v + 4];
}

// ---------------- init / prep ----------------
__global__ void init_kernel(float* agg, float* x2acc, float* emax, float* denom,
                            float* gsum, float* cnt) {
  int i = blockIdx.x * blockDim.x + threadIdx.x;
  if (i < NNODE * 64) { agg[i] = 0.f; x2acc[i] = 0.f; }
  if (i < NNODE) { emax[i] = __uint_as_float(0xFF800000u); denom[i] = 0.f; }
  if (i < NGRAPH * 64) gsum[i] = 0.f;
  if (i < NGRAPH) cnt[i] = 0.f;
}

// xb = bf16(x);  w3t[o*4096 + i*64 + k] = bf16(mlp_w3[k, i*64+o])  (K-major reindex)
__global__ void prep_kernel(const float* __restrict__ x, const float* __restrict__ w3,
                            unsigned short* __restrict__ xb, unsigned short* __restrict__ w3t) {
  int i = blockIdx.x * blockDim.x + threadIdx.x;
  if (i < NNODE * 64) xb[i] = f2bf(x[i]);
  if (i < 64 * 4096) {
    int o = i >> 12, K = i & 4095;
    int ii = K >> 6, k = K & 63;
    w3t[i] = f2bf(w3[k * 4096 + ii * 64 + o]);
  }
}

// ---------------- K1: edge MLP  h2 = relu(relu(ea@W1+b1)@W2+b2) -> bf16 ----------------
__global__ __launch_bounds__(256) void edge_mlp_kernel(
    const float* __restrict__ ea, const float* __restrict__ w1, const float* __restrict__ b1,
    const float* __restrict__ w2, const float* __restrict__ b2,
    unsigned short* __restrict__ h2g) {
  __shared__ unsigned short sW1[128 * 32];       // [n][k] transposed, 8KB
  __shared__ unsigned short sW2[64 * 128];       // [n][k] transposed, 16KB
  __shared__ unsigned short sH1[8 * 16 * 128];   // per-wave h1 tiles, 32KB
  int tid = threadIdx.x, wv = tid >> 5, lane = tid & 31;
  for (int i = tid; i < 32 * 128; i += 256) {
    int k = i >> 7, n = i & 127;
    sW1[n * 32 + k] = f2bf(w1[i]);
  }
  for (int i = tid; i < 128 * 64; i += 256) {
    int k = i >> 6, n = i & 63;
    sW2[n * 128 + k] = f2bf(w2[i]);
  }
  __syncthreads();

  int e0 = blockIdx.x * 128 + wv * 16;
  int n_ = lane & 15, rsel = (lane < 16) ? 0 : 8;
  unsigned a[8];
  a_frag_gf32(ea + e0 * FE, FE, 0, a);      // K = F_E = 32, one WMMA per tile
  unsigned short* h1w = sH1 + wv * 16 * 128;
#pragma unroll
  for (int nt = 0; nt < 8; ++nt) {
    unsigned b[8];
    b_frag_lds(sW1 + nt * 16 * 32, 32, 0, b);
    v8f c = {};
    c = wmma_bf16(a, b, c);
    int col = nt * 16 + n_;
    float bias = b1[col];
#pragma unroll
    for (int d = 0; d < 8; ++d) {
      float v = c[d] + bias;
      v = v > 0.f ? v : 0.f;
      h1w[(d + rsel) * 128 + col] = f2bf(v);
    }
  }
  __syncthreads();

  unsigned a4[4][8];
#pragma unroll
  for (int ks = 0; ks < 4; ++ks) a_frag_lds(h1w, 128, ks * 32, a4[ks]);
#pragma unroll
  for (int nt = 0; nt < 4; ++nt) {
    v8f c = {};
#pragma unroll
    for (int ks = 0; ks < 4; ++ks) {
      unsigned b[8];
      b_frag_lds(sW2 + nt * 16 * 128, 128, ks * 32, b);
      c = wmma_bf16(a4[ks], b, c);
    }
    int col = nt * 16 + n_;
    float bias = b2[col];
#pragma unroll
    for (int d = 0; d < 8; ++d) {
      float v = c[d] + bias;
      v = v > 0.f ? v : 0.f;
      h2g[(e0 + d + rsel) * 64 + col] = f2bf(v);
    }
  }
}

// ------- K2: fused NNConv bilinear GEMM  msg = Z @ W3flat,  Z = outer(x[src], h2) -------
__global__ __launch_bounds__(256) void nnconv_kernel(
    const unsigned short* __restrict__ h2g, const unsigned short* __restrict__ xb,
    const unsigned short* __restrict__ w3t, const float* __restrict__ b3,
    const int* __restrict__ eidx, float* __restrict__ agg) {
  __shared__ unsigned short sW3[64 * 256];     // W3flat chunk (4 i's), [n][256], 32KB
  __shared__ unsigned short sX[8 * 16 * 64];   // gathered x rows (bf16), 16KB
  int tid = threadIdx.x, wv = tid >> 5, lane = tid & 31;
  int e0 = blockIdx.x * 128 + wv * 16;
  __builtin_prefetch(eidx + NEDGE + e0, 0, 1);   // dst list used in epilogue
  unsigned short* sXw = sX + wv * 16 * 64;
  {  // gather x[src] rows: 2 lanes per row, 16 dwords each
    int r = lane >> 1, p = lane & 1;
    int srcn = eidx[e0 + r];
    const uint4* src = (const uint4*)(xb + srcn * 64 + p * 32);
    uint4* dst = (uint4*)(sXw + r * 64 + p * 32);
#pragma unroll
    for (int j = 0; j < 4; ++j) dst[j] = src[j];
  }
  // h2 row fragments (unscaled), A-layout pattern, loaded once
  unsigned hw[16];
  {
    int m = lane & 15, khi = (lane >> 4) * 8;
    const unsigned short* row = h2g + (e0 + m) * 64;
#pragma unroll
    for (int t = 0; t < 16; ++t) {
      int ks = t >> 3, v = t & 7;
      int k = ks * 32 + ((v < 4) ? 0 : 16) + khi + (v & 3) * 2;
      hw[t] = *(const unsigned*)(row + k);
    }
  }
  v8f zero = {};
  v8f acc[4];
#pragma unroll
  for (int nt = 0; nt < 4; ++nt) acc[nt] = zero;
  int m = lane & 15;
  unsigned sW3_off = (unsigned)(uintptr_t)(void*)sW3;   // LDS byte offset (addr[31:0])

#pragma unroll 1
  for (int ic = 0; ic < 16; ++ic) {       // 16 chunks x 4 i's = i in 0..63
    __syncthreads();                      // all waves done with previous chunk
    if (wv == 0) {
      // TDM: DMA [64 rows][256 elems] bf16 tile of W3flat into LDS (one per block)
      tdm_load_2d_bf16(sW3_off, w3t + ic * 256, 64, 4096, 256, 4096);
      __builtin_amdgcn_s_wait_tensorcnt(0);
    }
    __syncthreads();                      // publish DMA'd chunk to all waves
#pragma unroll 1
    for (int il = 0; il < 4; ++il) {
      int i = ic * 4 + il;
      unsigned sb = (unsigned)sXw[m * 64 + i];
      unsigned sp = sb | (sb << 16);           // broadcast scalar x[src,i] as bf16x2
      unsigned afr[2][8];
#pragma unroll
      for (int t = 0; t < 16; ++t) afr[t >> 3][t & 7] = pk_mul_bf16(hw[t], sp);
#pragma unroll
      for (int nt = 0; nt < 4; ++nt) {
        unsigned b[8];
        b_frag_lds(sW3 + nt * 16 * 256, 256, il * 64, b);
        acc[nt] = wmma_bf16(afr[0], b, acc[nt]);
        b_frag_lds(sW3 + nt * 16 * 256, 256, il * 64 + 32, b);
        acc[nt] = wmma_bf16(afr[1], b, acc[nt]);
      }
    }
  }
  // bias term: msg += x_src @ B3,  B3[i][o] = mlp_b3[i*64+o]   (reuse sW3 region)
  __syncthreads();
  for (int q = tid; q < 64 * 64; q += 256) {
    int n = q >> 6, i = q & 63;
    sW3[n * 64 + i] = f2bf(b3[i * 64 + n]);
  }
  __syncthreads();
  {
    unsigned ax[2][8];
    a_frag_lds(sXw, 64, 0, ax[0]);
    a_frag_lds(sXw, 64, 32, ax[1]);
#pragma unroll
    for (int nt = 0; nt < 4; ++nt) {
      unsigned b[8];
      b_frag_lds(sW3 + nt * 16 * 64, 64, 0, b);
      acc[nt] = wmma_bf16(ax[0], b, acc[nt]);
      b_frag_lds(sW3 + nt * 16 * 64, 64, 32, b);
      acc[nt] = wmma_bf16(ax[1], b, acc[nt]);
    }
  }
  // scatter-add messages to agg[dst]
  int n_ = lane & 15, rsel = (lane < 16) ? 0 : 8;
#pragma unroll
  for (int d = 0; d < 8; ++d) {
    int row = d + rsel;
    int dstn = eidx[NEDGE + e0 + row];
#pragma unroll
    for (int nt = 0; nt < 4; ++nt)
      atomicAdd(&agg[dstn * 64 + nt * 16 + n_], acc[nt][d]);
  }
}

// ---------------- K3/K4: node GEMM  out = [relu](in@W + bias [+ addend]) ----------------
__global__ __launch_bounds__(256) void node_gemm_kernel(
    const unsigned short* __restrict__ inb, const float* __restrict__ w,
    const float* __restrict__ bias, const float* __restrict__ addend,
    float* __restrict__ outf, unsigned short* __restrict__ outb, int do_relu) {
  __shared__ unsigned short sIn[128 * 64];   // 16KB
  __shared__ unsigned short sW[64 * 64];     // 8KB [o][i]
  int tid = threadIdx.x, wv = tid >> 5, lane = tid & 31;
  int r0 = blockIdx.x * 128;
  for (int q = tid; q < 64 * 64; q += 256) {
    int i = q >> 6, o = q & 63;
    sW[o * 64 + i] = f2bf(w[q]);
  }
  for (int q = tid * 8; q < 128 * 64; q += 256 * 8)
    *(uint4*)(sIn + q) = *(const uint4*)(inb + r0 * 64 + q);
  __syncthreads();

  const unsigned short* sInw = sIn + wv * 16 * 64;
  unsigned a2[2][8];
  a_frag_lds(sInw, 64, 0, a2[0]);
  a_frag_lds(sInw, 64, 32, a2[1]);
  int n_ = lane & 15, rsel = (lane < 16) ? 0 : 8;
  int rbase = r0 + wv * 16;
#pragma unroll
  for (int nt = 0; nt < 4; ++nt) {
    v8f c = {};
    unsigned b[8];
    b_frag_lds(sW + nt * 16 * 64, 64, 0, b);
    c = wmma_bf16(a2[0], b, c);
    b_frag_lds(sW + nt * 16 * 64, 64, 32, b);
    c = wmma_bf16(a2[1], b, c);
    int col = nt * 16 + n_;
    float bs = bias ? bias[col] : 0.f;
#pragma unroll
    for (int d = 0; d < 8; ++d) {
      int row = rbase + d + rsel;
      float v = c[d] + bs;
      if (addend) v += addend[row * 64 + col];
      if (do_relu) v = v > 0.f ? v : 0.f;
      if (outf) outf[row * 64 + col] = v;
      if (outb) outb[row * 64 + col] = f2bf(v);
    }
  }
}

// ---------------- GAT attention ----------------
__global__ void att_kernel(const float* __restrict__ hg, const float* __restrict__ asv,
                           const float* __restrict__ adv, float* as_, float* ad_) {
  int n = blockIdx.x * blockDim.x + threadIdx.x;
  if (n >= NNODE) return;
  float s = 0.f, d = 0.f;
  for (int c = 0; c < 64; ++c) {
    float h = hg[n * 64 + c];
    s += h * asv[c];
    d += h * adv[c];
  }
  as_[n] = s;
  ad_[n] = d;
}

__device__ __forceinline__ void atomicMaxF(float* addr, float val) {
  if (val >= 0.f) atomicMax((int*)addr, __float_as_int(val));
  else            atomicMin((unsigned int*)addr, __float_as_uint(val));
}

__global__ void gat1_kernel(const int* __restrict__ eidx, const float* __restrict__ as_,
                            const float* __restrict__ ad_, float* ebuf, float* emax) {
  int t = blockIdx.x * blockDim.x + threadIdx.x;
  if (t >= NEDGE + NNODE) return;
  int s = (t < NEDGE) ? eidx[t] : (t - NEDGE);
  int d = (t < NEDGE) ? eidx[NEDGE + t] : (t - NEDGE);
  float v = as_[s] + ad_[d];
  v = (v > 0.f) ? v : 0.2f * v;   // leaky relu
  ebuf[t] = v;
  atomicMaxF(&emax[d], v);
}

__global__ void gat2_kernel(const int* __restrict__ eidx, float* ebuf,
                            const float* __restrict__ emax, float* denom) {
  int t = blockIdx.x * blockDim.x + threadIdx.x;
  if (t >= NEDGE + NNODE) return;
  int d = (t < NEDGE) ? eidx[NEDGE + t] : (t - NEDGE);
  float ee = __expf(ebuf[t] - emax[d]);
  ebuf[t] = ee;
  atomicAdd(&denom[d], ee);
}

__global__ void gat3_kernel(const int* __restrict__ eidx, const float* __restrict__ ebuf,
                            const float* __restrict__ denom, const float* __restrict__ hg,
                            float* x2acc) {
  int q = blockIdx.x * blockDim.x + threadIdx.x;
  int t = q >> 2, part = q & 3;
  if (t >= NEDGE + NNODE) return;
  int s = (t < NEDGE) ? eidx[t] : (t - NEDGE);
  int d = (t < NEDGE) ? eidx[NEDGE + t] : (t - NEDGE);
  float alpha = ebuf[t] / denom[d];
  const float* hr = hg + s * 64 + part * 16;
  float* ar = x2acc + d * 64 + part * 16;
#pragma unroll
  for (int c = 0; c < 16; ++c) atomicAdd(&ar[c], alpha * hr[c]);
}

__global__ void pool_kernel(const float* __restrict__ x2acc, const float* __restrict__ gatb,
                            const int* __restrict__ batch, float* gsum, float* cnt) {
  int n = blockIdx.x * blockDim.x + threadIdx.x;
  if (n >= NNODE) return;
  int b = batch[n];
  for (int c = 0; c < 64; ++c) {
    float v = x2acc[n * 64 + c] + gatb[c];
    v = v > 0.f ? v : 0.f;
    atomicAdd(&gsum[b * 64 + c], v);
  }
  atomicAdd(&cnt[b], 1.f);
}

// ---------------- dense head (tiny: ~2.4 MFLOP) ----------------
__global__ __launch_bounds__(256) void head_kernel(
    const float* __restrict__ gsum, const float* __restrict__ cnt,
    const float* __restrict__ w1, const float* __restrict__ b1,
    const float* __restrict__ w2, const float* __restrict__ b2,
    const float* __restrict__ w3, const float* __restrict__ b3, float* __restrict__ out) {
  __shared__ float bufA[64 * 64];
  __shared__ float bufB[64 * 128];
  int tid = threadIdx.x;
  for (int q = tid; q < 64 * 64; q += 256) {
    int g = q >> 6;
    float c = cnt[g];
    c = c > 1.f ? c : 1.f;
    bufA[q] = gsum[q] / c;
  }
  __syncthreads();
  for (int q = tid; q < 64 * 128; q += 256) {
    int r = q >> 7, c = q & 127;
    float s = b1[c];
    for (int k = 0; k < 64; ++k) s += bufA[r * 64 + k] * w1[k * 128 + c];
    bufB[q] = s > 0.f ? s : 0.f;
  }
  __syncthreads();
  for (int q = tid; q < 64 * 64; q += 256) {
    int r = q >> 6, c = q & 63;
    float s = b2[c];
    for (int k = 0; k < 128; ++k) s += bufB[r * 128 + k] * w2[k * 64 + c];
    bufA[q] = s > 0.f ? s : 0.f;
  }
  __syncthreads();
  for (int q = tid; q < 64 * 32; q += 256) {
    int r = q >> 5, c = q & 31;
    float s = b3[c];
    for (int k = 0; k < 64; ++k) s += bufA[r * 64 + k] * w3[k * 32 + c];
    out[q] = s > 0.f ? s : 0.f;
  }
}

extern "C" void kernel_launch(void* const* d_in, const int* in_sizes, int n_in,
                              void* d_out, int out_size, void* d_ws, size_t ws_size,
                              hipStream_t stream) {
  (void)in_sizes; (void)n_in; (void)out_size; (void)ws_size;
  const float* x       = (const float*)d_in[0];
  const float* eattr   = (const float*)d_in[1];
  const float* mlp_w1  = (const float*)d_in[2];
  const float* mlp_b1  = (const float*)d_in[3];
  const float* mlp_w2  = (const float*)d_in[4];
  const float* mlp_b2  = (const float*)d_in[5];
  const float* mlp_w3  = (const float*)d_in[6];
  const float* mlp_b3  = (const float*)d_in[7];
  const float* root_w  = (const float*)d_in[8];
  const float* conv1_b = (const float*)d_in[9];
  const float* gat_w   = (const float*)d_in[10];
  const float* att_src = (const float*)d_in[11];
  const float* att_dst = (const float*)d_in[12];
  const float* gat_b   = (const float*)d_in[13];
  const float* fc1_w   = (const float*)d_in[14];
  const float* fc1_b   = (const float*)d_in[15];
  const float* fc2_w   = (const float*)d_in[16];
  const float* fc2_b   = (const float*)d_in[17];
  const float* fc3_w   = (const float*)d_in[18];
  const float* fc3_b   = (const float*)d_in[19];
  const int* eidx      = (const int*)d_in[20];
  const int* batch     = (const int*)d_in[21];

  char* p = (char*)d_ws;
  auto alloc = [&](size_t bytes) -> char* {
    char* r = p;
    p += (bytes + 255) & ~(size_t)255;
    return r;
  };
  unsigned short* h2b = (unsigned short*)alloc((size_t)NEDGE * 64 * 2);
  unsigned short* w3t = (unsigned short*)alloc((size_t)64 * 4096 * 2);
  unsigned short* xb  = (unsigned short*)alloc((size_t)NNODE * 64 * 2);
  unsigned short* x1b = (unsigned short*)alloc((size_t)NNODE * 64 * 2);
  float* agg   = (float*)alloc((size_t)NNODE * 64 * 4);
  float* hg    = (float*)alloc((size_t)NNODE * 64 * 4);
  float* as_   = (float*)alloc((size_t)NNODE * 4);
  float* ad_   = (float*)alloc((size_t)NNODE * 4);
  float* ebuf  = (float*)alloc((size_t)(NEDGE + NNODE) * 4);
  float* emax  = (float*)alloc((size_t)NNODE * 4);
  float* denom = (float*)alloc((size_t)NNODE * 4);
  float* x2acc = (float*)alloc((size_t)NNODE * 64 * 4);
  float* gsum  = (float*)alloc((size_t)NGRAPH * 64 * 4);
  float* cnt   = (float*)alloc((size_t)NGRAPH * 4);

  init_kernel<<<(NNODE * 64 + 255) / 256, 256, 0, stream>>>(agg, x2acc, emax, denom, gsum, cnt);
  prep_kernel<<<(NNODE * 64 + 255) / 256, 256, 0, stream>>>(x, mlp_w3, xb, w3t);
  edge_mlp_kernel<<<NEDGE / 128, 256, 0, stream>>>(eattr, mlp_w1, mlp_b1, mlp_w2, mlp_b2, h2b);
  nnconv_kernel<<<NEDGE / 128, 256, 0, stream>>>(h2b, xb, w3t, mlp_b3, eidx, agg);
  // x1 = relu(agg + x@root_w + conv1_b)  (bf16 out for next GEMM)
  node_gemm_kernel<<<NNODE / 128, 256, 0, stream>>>(xb, root_w, conv1_b, agg,
                                                    (float*)nullptr, x1b, 1);
  // hg = x1 @ gat_w  (no relu, no bias)
  node_gemm_kernel<<<NNODE / 128, 256, 0, stream>>>(x1b, gat_w, (const float*)nullptr,
                                                    (const float*)nullptr, hg,
                                                    (unsigned short*)nullptr, 0);
  att_kernel<<<NNODE / 256, 256, 0, stream>>>(hg, att_src, att_dst, as_, ad_);
  gat1_kernel<<<(NEDGE + NNODE + 255) / 256, 256, 0, stream>>>(eidx, as_, ad_, ebuf, emax);
  gat2_kernel<<<(NEDGE + NNODE + 255) / 256, 256, 0, stream>>>(eidx, ebuf, emax, denom);
  gat3_kernel<<<((NEDGE + NNODE) * 4 + 255) / 256, 256, 0, stream>>>(eidx, ebuf, denom, hg,
                                                                     x2acc);
  pool_kernel<<<NNODE / 256, 256, 0, stream>>>(x2acc, gat_b, batch, gsum, cnt);
  head_kernel<<<1, 256, 0, stream>>>(gsum, cnt, fc1_w, fc1_b, fc2_w, fc2_b, fc3_w, fc3_b,
                                     (float*)d_out);
}
